// GridNetBlock_Att_53274774339837
// MI455X (gfx1250) — compile-verified, matrix operand order checked by reading, not run on hardware
//
#include <hip/hip_runtime.h>
#include <math.h>

typedef __attribute__((ext_vector_type(16))) _Float16 v16h;
typedef __attribute__((ext_vector_type(8)))  float    v8f;
typedef _Float16 h16;

// ---------------- problem constants ----------------
constexpr int B_ = 4, C_ = 64, T_ = 1000, F_ = 129;
constexpr int H_ = 4, E_ = 4, D_ = 16, K_ = 3, GROUP_ = 4;
constexpr int TF_   = T_ * F_;           // 129000
constexpr int CTF_  = C_ * TF_;          // 8,256,000
constexpr int NTF_  = B_ * TF_;          // 516,000
constexpr long long NCTF_ = (long long)B_ * CTF_;   // 33,024,000
constexpr int EFV_ = E_ * F_;            // 516 valid Q/K row
constexpr int EFP_ = 544;                // padded to 17*32 (1088B rows, 32B aligned)
constexpr int DF_  = D_ * F_;            // 2064
constexpr int DFP_ = 2080;               // V-transposed rows padded (2064+16)
constexpr int TP_  = 1024;               // padded T
constexpr long long PSZ_ = (long long)TP_ * TP_;    // per (h,b) score tile
constexpr float EPSF = 1e-5f;

// ---------------- ws layout (float units; f16 regions counted /2) ----------------
constexpr size_t OFF_BSUM = 0;
constexpr size_t OFF_TMP  = (size_t)NCTF_;            // b1 -> y -> o_buf
constexpr size_t OFF_OUT  = 2 * (size_t)NCTF_;        // "output" tensor
constexpr size_t QH_HALF  = (size_t)H_ * B_ * TP_ * EFP_;   // 8,912,896 halves
constexpr size_t OFF_Q    = 3 * (size_t)NCTF_;              // f16
constexpr size_t OFF_K    = OFF_Q + QH_HALF / 2;            // f16
constexpr size_t VM_F32   = (size_t)H_ * B_ * TP_ * DF_;    // 33,816,576 floats
constexpr size_t OFF_VM   = OFF_K + QH_HALF / 2;            // f32 (t-major V)
constexpr size_t VT_HALF  = (size_t)H_ * B_ * DFP_ * TP_;   // 34,078,720 halves
constexpr size_t OFF_VT   = OFF_VM + VM_F32;                // f16 (n-major V)
constexpr size_t OFF_PSC  = OFF_VT + VT_HALF / 2;           // f32 scores
constexpr size_t PALL_    = (size_t)(H_ * B_) * PSZ_;       // 16,777,216
constexpr size_t OFF_PP   = OFF_PSC + PALL_;                // f16 probs
constexpr size_t OFF_MU   = OFF_PP + PALL_ / 2;
constexpr size_t OFF_RSTD = OFF_MU + NTF_;
constexpr size_t OFF_GAPR = OFF_RSTD + NTF_;          // B*C*T = 256000
constexpr size_t OFF_GAPC = OFF_GAPR + (size_t)B_ * C_ * T_;
constexpr size_t OFF_R    = OFF_GAPC + (size_t)B_ * C_ * F_;
constexpr size_t OFF_FILT = OFF_R + 256;
constexpr size_t OFF_GS   = OFF_FILT + 64;
constexpr size_t OFF_GQ   = OFF_GS + 256;
constexpr size_t OFF_GN   = OFF_GQ + 256;

// ---------------- WMMA fragment helpers ----------------
template <typename GET>
__device__ inline v16h frag_a16x32(GET get) {
  int lane = threadIdx.x & 31;
  int row = lane & 15;
  int k0 = (lane >> 4) * 16;
  v16h a;
#pragma unroll
  for (int i = 0; i < 16; ++i) a[i] = (_Float16)get(row, k0 + i);
  return a;
}
template <typename GET>
__device__ inline v16h frag_b32x16(GET get) {
  int lane = threadIdx.x & 31;
  int col = lane & 15;
  int k0 = (lane >> 4) * 16;
  v16h b;
#pragma unroll
  for (int i = 0; i < 16; ++i) b[i] = (_Float16)get(k0 + i, col);
  return b;
}
template <typename PUT>
__device__ inline void frag_c_store(const v8f& c, PUT put) {
  int lane = threadIdx.x & 31;
  int n = lane & 15;
  int mb = (lane >> 4) * 8;
#pragma unroll
  for (int r = 0; r < 8; ++r) put(mb + r, n, c[r]);
}

// ---------------- block reductions (blockDim == 256) ----------------
__device__ inline void block_reduce2(float& a, float& b) {
  __shared__ float sa[256], sb[256];
  int tid = threadIdx.x;
  sa[tid] = a; sb[tid] = b; __syncthreads();
  for (int s = 128; s > 0; s >>= 1) {
    if (tid < s) { sa[tid] += sa[tid + s]; sb[tid] += sb[tid + s]; }
    __syncthreads();
  }
  a = sa[0]; b = sb[0]; __syncthreads();
}
__device__ inline float block_reduce_max(float v) {
  __shared__ float sm[256];
  int tid = threadIdx.x;
  sm[tid] = v; __syncthreads();
  for (int s = 128; s > 0; s >>= 1) {
    if (tid < s) sm[tid] = fmaxf(sm[tid], sm[tid + s]);
    __syncthreads();
  }
  float r = sm[0]; __syncthreads();
  return r;
}

// ---------------- LN over channel axis: per-(b,t,f) mu / rstd ----------------
__global__ void k_ln_stats(const float* __restrict__ X, float* __restrict__ mu,
                           float* __restrict__ rstd) {
  int idx = blockIdx.x * 256 + threadIdx.x;
  if (idx >= NTF_) return;
  int b = idx / TF_, pos = idx % TF_;
  const float* p = X + (size_t)b * CTF_ + pos;
  float s = 0.f, q = 0.f;
#pragma unroll 4
  for (int c = 0; c < C_; ++c) { float v = p[(size_t)c * TF_]; s += v; q += v * v; }
  float m = s * (1.f / C_);
  float var = q * (1.f / C_) - m * m;
  var = fmaxf(var, EPSF);
  mu[idx] = m;
  rstd[idx] = 1.f / sqrtf(var);
}

// ---------------- channel means of LN output: R[b,c] ----------------
__global__ void k_gapc(const float* __restrict__ X, const float* __restrict__ mu,
                       const float* __restrict__ rstd, const float* __restrict__ g,
                       const float* __restrict__ bb, float* __restrict__ R) {
  int blk = blockIdx.x;                 // b*C + c
  int b = blk / C_, c = blk % C_;
  const float* p  = X + (size_t)blk * TF_;
  const float* mp = mu + (size_t)b * TF_;
  const float* rp = rstd + (size_t)b * TF_;
  float s = 0.f, q = 0.f;
  for (int i = threadIdx.x; i < TF_; i += 256) s += (p[i] - mp[i]) * rp[i];
  block_reduce2(s, q);
  if (threadIdx.x == 0) R[blk] = g[c] * (s / (float)TF_) + bb[c];
}

// ---------------- filt[b,12] = tanh(R @ W^T) ----------------
__global__ void k_filt(const float* __restrict__ R, const float* __restrict__ W,
                       float* __restrict__ filt) {
  int tid = threadIdx.x;
  if (tid >= B_ * GROUP_ * K_) return;
  int n = tid / (GROUP_ * K_), j = tid % (GROUP_ * K_);
  float s = 0.f;
  for (int c = 0; c < C_; ++c) s += R[n * C_ + c] * W[j * C_ + c];
  filt[tid] = tanhf(s);
}

// ---------------- row mean of LN output along F: gap[b,c,t] ----------------
__global__ void k_rowmean(const float* __restrict__ X, const float* __restrict__ mu,
                          const float* __restrict__ rstd, const float* __restrict__ g,
                          const float* __restrict__ bb, float* __restrict__ gapr) {
  int idx = blockIdx.x * 256 + threadIdx.x;
  if (idx >= B_ * C_ * T_) return;
  int t = idx % T_, c = (idx / T_) % C_, b = idx / (C_ * T_);
  const float* xp = X + ((size_t)(b * C_ + c) * T_ + t) * F_;
  const float* mp = mu + (size_t)b * TF_ + (size_t)t * F_;
  const float* rp = rstd + (size_t)b * TF_ + (size_t)t * F_;
  float s = 0.f;
  for (int f = 0; f < F_; ++f) s += (xp[f] - mp[f]) * rp[f];
  gapr[idx] = g[c] * (s / (float)F_) + bb[c];
}

// ---------------- column mean of LN output along T: gap[b,c,f] ----------------
__global__ void k_colmean(const float* __restrict__ X, const float* __restrict__ mu,
                          const float* __restrict__ rstd, const float* __restrict__ g,
                          const float* __restrict__ bb, float* __restrict__ gapc) {
  int idx = blockIdx.x * 256 + threadIdx.x;
  if (idx >= B_ * C_ * F_) return;
  int f = idx % F_, c = (idx / F_) % C_, b = idx / (C_ * F_);
  const float* xp = X + (size_t)(b * C_ + c) * TF_ + f;
  const float* mp = mu + (size_t)b * TF_ + f;
  const float* rp = rstd + (size_t)b * TF_ + f;
  float s = 0.f;
  for (int t = 0; t < T_; ++t)
    s += (xp[(size_t)t * F_] - mp[(size_t)t * F_]) * rp[(size_t)t * F_];
  gapc[idx] = g[c] * (s / (float)T_) + bb[c];
}

// ---------------- LISA horizontal (axis = F, reflect pad) ----------------
__global__ void k_lisa_h(const float* __restrict__ X, const float* __restrict__ mu,
                         const float* __restrict__ rstd, const float* __restrict__ g,
                         const float* __restrict__ bb, const float* __restrict__ gapr,
                         const float* __restrict__ filt, const float* __restrict__ ia,
                         const float* __restrict__ ll, const float* __restrict__ lh,
                         int dil, float* __restrict__ OUT) {
  long long idx = (long long)blockIdx.x * 256 + threadIdx.x;
  if (idx >= NCTF_) return;
  int f = (int)(idx % F_);
  int t = (int)((idx / F_) % T_);
  int c = (int)((idx / TF_) % C_);
  int b = (int)(idx / CTF_);
  const float* xp = X + ((size_t)(b * C_ + c) * T_ + t) * F_;
  const float* mp = mu + (size_t)b * TF_ + (size_t)t * F_;
  const float* rp = rstd + (size_t)b * TF_ + (size_t)t * F_;
  float gc = g[c], bc = bb[c];
  int fm = f - dil; fm = fm < 0 ? -fm : fm;
  int fp = f + dil; fp = fp >= F_ ? 2 * F_ - 2 - fp : fp;
  float y0 = (xp[fm] - mp[fm]) * rp[fm] * gc + bc;
  float y1 = (xp[f]  - mp[f])  * rp[f]  * gc + bc;
  float y2 = (xp[fp] - mp[fp]) * rp[fp] * gc + bc;
  const float* fl = filt + b * (GROUP_ * K_) + (c >> 4) * K_;
  float out = fl[0] * y0 + fl[1] * y1 + fl[2] * y2;
  float iac = ia[c];
  float res = (out * (iac + 1.f) - iac * gapr[(size_t)(b * C_ + c) * T_ + t]) * ll[c]
            + y1 * (lh[c] + 1.f);
  OUT[idx] = res;
}

// ---------------- LISA vertical (axis = T) + mix accumulate into bsum ----------------
__global__ void k_lisa_v_acc(const float* __restrict__ X, const float* __restrict__ mu,
                             const float* __restrict__ rstd, const float* __restrict__ g,
                             const float* __restrict__ bb, const float* __restrict__ gapc,
                             const float* __restrict__ filt, const float* __restrict__ ia,
                             const float* __restrict__ ll, const float* __restrict__ lh,
                             int dil, const float* __restrict__ Xorig,
                             const float* __restrict__ mixg, const float* __restrict__ mixb,
                             float* __restrict__ bsum, int init) {
  long long idx = (long long)blockIdx.x * 256 + threadIdx.x;
  if (idx >= NCTF_) return;
  int f = (int)(idx % F_);
  int t = (int)((idx / F_) % T_);
  int c = (int)((idx / TF_) % C_);
  int b = (int)(idx / CTF_);
  const float* xp = X + (size_t)(b * C_ + c) * TF_ + f;
  const float* mp = mu + (size_t)b * TF_ + f;
  const float* rp = rstd + (size_t)b * TF_ + f;
  float gc = g[c], bc = bb[c];
  int tm = t - dil; tm = tm < 0 ? -tm : tm;
  int tp = t + dil; tp = tp >= T_ ? 2 * T_ - 2 - tp : tp;
  float y0 = (xp[(size_t)tm * F_] - mp[(size_t)tm * F_]) * rp[(size_t)tm * F_] * gc + bc;
  float y1 = (xp[(size_t)t  * F_] - mp[(size_t)t  * F_]) * rp[(size_t)t  * F_] * gc + bc;
  float y2 = (xp[(size_t)tp * F_] - mp[(size_t)tp * F_]) * rp[(size_t)tp * F_] * gc + bc;
  const float* fl = filt + b * (GROUP_ * K_) + (c >> 4) * K_;
  float out = fl[0] * y0 + fl[1] * y1 + fl[2] * y2;
  float iac = ia[c];
  float res = (out * (iac + 1.f) - iac * gapc[(size_t)(b * C_ + c) * F_ + f]) * ll[c]
            + y1 * (lh[c] + 1.f);
  float v = mixg[c] * res + Xorig[idx] * mixb[c];
  if (init) bsum[idx] = v;
  else      bsum[idx] += v;
}

// ---------------- 64x64 channel GEMM via WMMA (unguarded loads) ----------------
__global__ void k_gemm_cc(const float* __restrict__ W, const float* __restrict__ X,
                          const float* __restrict__ bias, float* __restrict__ Y,
                          int fuse_prelu, const float* __restrict__ alpha) {
  int n0 = blockIdx.x * 16, m0 = blockIdx.y * 16, b = blockIdx.z;
  const float* Xb = X + (size_t)b * CTF_;
  float* Yb = Y + (size_t)b * CTF_;
  v8f c;
#pragma unroll
  for (int r = 0; r < 8; ++r) c[r] = 0.f;
#pragma unroll
  for (int k0 = 0; k0 < C_; k0 += 32) {
    v16h a = frag_a16x32([&](int m, int k) { return W[(m0 + m) * C_ + k0 + k]; });
    v16h bf = frag_b32x16([&](int k, int n) {
      return Xb[(size_t)(k0 + k) * TF_ + n0 + n];
    });
    c = __builtin_amdgcn_wmma_f32_16x16x32_f16(false, a, false, bf, (short)0, c,
                                               false, false);
  }
  float aph = fuse_prelu ? alpha[0] : 0.f;
  frag_c_store(c, [&](int m, int n, float v) {
    int nn = n0 + n;
    if (nn < TF_) {
      v += bias[m0 + m];
      if (fuse_prelu && v < 0.f) v *= aph;
      Yb[(size_t)(m0 + m) * TF_ + nn] = v;
    }
  });
}

// ---------------- per-batch norm (groupnorm over C,T,F) ----------------
__global__ void k_gn_part(const float* __restrict__ Y, float* __restrict__ pS,
                          float* __restrict__ pQ) {
  int blk = blockIdx.x;  // b*C + c
  const float* p = Y + (size_t)blk * TF_;
  float s = 0.f, q = 0.f;
  for (int i = threadIdx.x; i < TF_; i += 256) { float v = p[i]; s += v; q += v * v; }
  block_reduce2(s, q);
  if (threadIdx.x == 0) { pS[blk] = s; pQ[blk] = q; }
}
__global__ void k_gn_fin(const float* __restrict__ pS, const float* __restrict__ pQ,
                         float* __restrict__ st) {
  __shared__ float sa[64], sb[64];
  int b = blockIdx.x, tid = threadIdx.x;
  sa[tid] = pS[b * C_ + tid]; sb[tid] = pQ[b * C_ + tid]; __syncthreads();
  for (int s = 32; s > 0; s >>= 1) {
    if (tid < s) { sa[tid] += sa[tid + s]; sb[tid] += sb[tid + s]; }
    __syncthreads();
  }
  if (tid == 0) {
    float n = (float)C_ * (float)TF_;
    float m = sa[0] / n;
    float var = sb[0] / n - m * m;
    st[b * 2] = m;
    st[b * 2 + 1] = 1.f / sqrtf(var + EPSF);
  }
}
__global__ void k_gn_apply(const float* __restrict__ Y, const float* __restrict__ st,
                           const float* __restrict__ gg, const float* __restrict__ gb,
                           const float* __restrict__ alpha, float* __restrict__ OUT) {
  long long idx = (long long)blockIdx.x * 256 + threadIdx.x;
  if (idx >= NCTF_) return;
  int c = (int)((idx / TF_) % C_);
  int b = (int)(idx / CTF_);
  float v = (Y[idx] - st[b * 2]) * st[b * 2 + 1] * gg[c] + gb[c];
  float a = alpha[0];
  OUT[idx] = v < 0.f ? a * v : v;
}

// ---------------- head projections (f16 out, Q/K): pad rows/cols zeroed --------
__global__ void k_headproj_h(const float* __restrict__ OUT, const float* __restrict__ w,
                             const float* __restrict__ bias, const float* __restrict__ alpha,
                             h16* __restrict__ Z, int Ee, int stride) {
  long long idx = (long long)blockIdx.x * 256 + threadIdx.x;
  long long total = (long long)H_ * B_ * TP_ * stride;
  if (idx >= total) return;
  int j = (int)(idx % stride);
  int t = (int)((idx / stride) % TP_);
  int b = (int)((idx / ((long long)stride * TP_)) % B_);
  int h = (int)(idx / ((long long)stride * TP_ * B_));
  int valid = Ee * F_;
  float val = 0.f;
  if (j < valid && t < T_) {
    int e = j / F_, f = j % F_;
    const float* xp = OUT + ((size_t)b * C_ * T_ + t) * F_ + f;
    const float* wp = w + (h * Ee + e) * C_;
    float acc = bias[h * Ee + e];
#pragma unroll 4
    for (int c = 0; c < C_; ++c) acc += xp[(size_t)c * TF_] * wp[c];
    float a = alpha[h];
    val = acc < 0.f ? a * acc : acc;
  }
  Z[idx] = (h16)val;
}

// ---------------- head projection (f32 out, V t-major, pad rows zero) ----------
__global__ void k_headproj_f(const float* __restrict__ OUT, const float* __restrict__ w,
                             const float* __restrict__ bias, const float* __restrict__ alpha,
                             float* __restrict__ Z, int Ee, int stride) {
  long long idx = (long long)blockIdx.x * 256 + threadIdx.x;
  long long total = (long long)H_ * B_ * TP_ * stride;
  if (idx >= total) return;
  int j = (int)(idx % stride);
  int t = (int)((idx / stride) % TP_);
  int b = (int)((idx / ((long long)stride * TP_)) % B_);
  int h = (int)(idx / ((long long)stride * TP_ * B_));
  int valid = Ee * F_;
  float val = 0.f;
  if (j < valid && t < T_) {
    int e = j / F_, f = j % F_;
    const float* xp = OUT + ((size_t)b * C_ * T_ + t) * F_ + f;
    const float* wp = w + (h * Ee + e) * C_;
    float acc = bias[h * Ee + e];
#pragma unroll 4
    for (int c = 0; c < C_; ++c) acc += xp[(size_t)c * TF_] * wp[c];
    float a = alpha[h];
    val = acc < 0.f ? a * acc : acc;
  }
  Z[idx] = val;
}

// ---------------- per-(h,b,t) row layernorm over (e,f), f16 rows ----------------
__global__ void k_rownorm_h(h16* __restrict__ Z, const float* __restrict__ lng,
                            const float* __restrict__ lnb, int Ee, int stride) {
  int blk = blockIdx.x;  // H*B*T valid rows
  int h = blk / (B_ * T_);
  int rem = blk % (B_ * T_);
  int b = rem / T_, t = rem % T_;
  h16* row = Z + ((size_t)(h * B_ + b) * TP_ + t) * stride;
  int valid = Ee * F_;
  float s = 0.f, q = 0.f;
  for (int j = threadIdx.x; j < valid; j += 256) {
    float v = (float)row[j]; s += v; q += v * v;
  }
  block_reduce2(s, q);
  float m = s / (float)valid;
  float var = q / (float)valid - m * m;
  float rs = 1.f / sqrtf(var + EPSF);
  for (int j = threadIdx.x; j < valid; j += 256) {
    int e = j / F_, f = j % F_;
    int li = (h * Ee + e) * F_ + f;
    row[j] = (h16)(((float)row[j] - m) * rs * lng[li] + lnb[li]);
  }
}

// ---------------- per-(h,b,t) row layernorm, f32 rows (V) ----------------
__global__ void k_rownorm_f(float* __restrict__ Z, const float* __restrict__ lng,
                            const float* __restrict__ lnb, int Ee, int stride) {
  int blk = blockIdx.x;
  int h = blk / (B_ * T_);
  int rem = blk % (B_ * T_);
  int b = rem / T_, t = rem % T_;
  float* row = Z + ((size_t)(h * B_ + b) * TP_ + t) * stride;
  int valid = Ee * F_;
  float s = 0.f, q = 0.f;
  for (int j = threadIdx.x; j < valid; j += 256) { float v = row[j]; s += v; q += v * v; }
  block_reduce2(s, q);
  float m = s / (float)valid;
  float var = q / (float)valid - m * m;
  float rs = 1.f / sqrtf(var + EPSF);
  for (int j = threadIdx.x; j < valid; j += 256) {
    int e = j / F_, f = j % F_;
    int li = (h * Ee + e) * F_ + f;
    row[j] = (row[j] - m) * rs * lng[li] + lnb[li];
  }
}

// ---------------- V transpose: (hb, s, n) f32 -> (hb, n, s) f16, n padded ------
__global__ void k_v_transpose(const float* __restrict__ Vm, h16* __restrict__ Vt) {
  long long idx = (long long)blockIdx.x * 256 + threadIdx.x;
  long long total = (long long)H_ * B_ * DFP_ * TP_;
  if (idx >= total) return;
  int s = (int)(idx % TP_);
  int n = (int)((idx / TP_) % DFP_);
  int hb = (int)(idx / ((long long)TP_ * DFP_));
  float val = (n < DF_) ? Vm[((size_t)hb * TP_ + s) * DF_ + n] : 0.f;
  Vt[idx] = (h16)val;
}

// ---------------- attention scores: 16x32 tile/wave, direct f16 fragments -------
__global__ void k_att_scores(const h16* __restrict__ Q, const h16* __restrict__ K,
                             float* __restrict__ P, float invscale) {
  int s0 = blockIdx.x * 32, t0 = blockIdx.y * 16, hb = blockIdx.z;
  const h16* Qh = Q + (size_t)hb * TP_ * EFP_;
  const h16* Kh = K + (size_t)hb * TP_ * EFP_;
  float* Ph = P + (size_t)hb * PSZ_;
  int lane = threadIdx.x & 31;
  int r = lane & 15;
  int koff = (lane >> 4) * 16;
  v8f c0, c1;
#pragma unroll
  for (int i = 0; i < 8; ++i) { c0[i] = 0.f; c1[i] = 0.f; }
  for (int k0 = 0; k0 < EFP_; k0 += 32) {
    v16h a  = *(const v16h*)(Qh + (size_t)(t0 + r) * EFP_ + k0 + koff);
    v16h b0 = *(const v16h*)(Kh + (size_t)(s0 + r) * EFP_ + k0 + koff);
    v16h b1 = *(const v16h*)(Kh + (size_t)(s0 + 16 + r) * EFP_ + k0 + koff);
    c0 = __builtin_amdgcn_wmma_f32_16x16x32_f16(false, a, false, b0, (short)0, c0,
                                                false, false);
    c1 = __builtin_amdgcn_wmma_f32_16x16x32_f16(false, a, false, b1, (short)0, c1,
                                                false, false);
  }
  frag_c_store(c0, [&](int m, int n, float v) {
    Ph[(size_t)(t0 + m) * TP_ + s0 + n] = v * invscale;
  });
  frag_c_store(c1, [&](int m, int n, float v) {
    Ph[(size_t)(t0 + m) * TP_ + s0 + 16 + n] = v * invscale;
  });
}

// ---------------- softmax: f32 scores -> f16 probs (pads zeroed) ----------------
__global__ void k_softmax(const float* __restrict__ Ps, h16* __restrict__ Pp) {
  __shared__ float ebuf[TP_];
  int blk = blockIdx.x;                    // hb*TP + t (all padded rows)
  int hb = blk / TP_, t = blk % TP_;
  h16* orow = Pp + (size_t)hb * PSZ_ + (size_t)t * TP_;
  if (t >= T_) {
    for (int s = threadIdx.x; s < TP_; s += 256) orow[s] = (h16)0.f;
    return;
  }
  const float* row = Ps + (size_t)hb * PSZ_ + (size_t)t * TP_;
  float mx = -3.4e38f;
  for (int s = threadIdx.x; s < T_; s += 256) mx = fmaxf(mx, row[s]);
  mx = block_reduce_max(mx);
  float sum = 0.f, dummy = 0.f;
  for (int s = threadIdx.x; s < T_; s += 256) {
    float e = expf(row[s] - mx);
    ebuf[s] = e;
    sum += e;
  }
  block_reduce2(sum, dummy);
  float inv = 1.f / sum;
  for (int s = threadIdx.x; s < TP_; s += 256)
    orow[s] = (h16)(s < T_ ? ebuf[s] * inv : 0.f);
}

// ---------------- attn @ V: 16x32 tile/wave, direct f16 fragments ----------------
__global__ void k_att_o(const h16* __restrict__ Pp, const h16* __restrict__ Vt,
                        float* __restrict__ O) {
  int n0 = blockIdx.x * 32, t0 = blockIdx.y * 16, hb = blockIdx.z;
  int h = hb / B_, b = hb % B_;
  const h16* Ph = Pp + (size_t)hb * PSZ_;
  const h16* Vh = Vt + (size_t)hb * DFP_ * TP_;
  int lane = threadIdx.x & 31;
  int r = lane & 15;
  int koff = (lane >> 4) * 16;
  v8f c0, c1;
#pragma unroll
  for (int i = 0; i < 8; ++i) { c0[i] = 0.f; c1[i] = 0.f; }
  for (int k0 = 0; k0 < TP_; k0 += 32) {
    v16h a  = *(const v16h*)(Ph + (size_t)(t0 + r) * TP_ + k0 + koff);
    v16h b0 = *(const v16h*)(Vh + (size_t)(n0 + r) * TP_ + k0 + koff);
    v16h b1 = *(const v16h*)(Vh + (size_t)(n0 + 16 + r) * TP_ + k0 + koff);
    c0 = __builtin_amdgcn_wmma_f32_16x16x32_f16(false, a, false, b0, (short)0, c0,
                                                false, false);
    c1 = __builtin_amdgcn_wmma_f32_16x16x32_f16(false, a, false, b1, (short)0, c1,
                                                false, false);
  }
  auto emit = [&](int m, int nn, float v) {
    int t = t0 + m;
    if (t >= T_ || nn >= DF_) return;
    int d = nn / F_, f = nn % F_;
    O[(((size_t)b * C_ + h * D_ + d) * T_ + t) * F_ + f] = v;
  };
  frag_c_store(c0, [&](int m, int n, float v) { emit(m, n0 + n, v); });
  frag_c_store(c1, [&](int m, int n, float v) { emit(m, n0 + 16 + n, v); });
}

// ---------------- final: ln4dcf over (c,f) per (b,t), + residual ----------------
__global__ void k_lncf_final(const float* __restrict__ Pb, const float* __restrict__ OUT,
                             const float* __restrict__ lng, const float* __restrict__ lnb,
                             float* __restrict__ dst) {
  int blk = blockIdx.x;  // b*T + t
  int b = blk / T_, t = blk % T_;
  float s = 0.f, q = 0.f;
  for (int j = threadIdx.x; j < C_ * F_; j += 256) {
    int c = j / F_, f = j % F_;
    float v = Pb[((size_t)(b * C_ + c) * T_ + t) * F_ + f];
    s += v; q += v * v;
  }
  block_reduce2(s, q);
  float n = (float)(C_ * F_);
  float m = s / n;
  float var = q / n - m * m;
  float rs = 1.f / sqrtf(var + EPSF);
  for (int j = threadIdx.x; j < C_ * F_; j += 256) {
    int c = j / F_, f = j % F_;
    size_t gi = ((size_t)(b * C_ + c) * T_ + t) * F_ + f;
    float v = (Pb[gi] - m) * rs * lng[c * F_ + f] + lnb[c * F_ + f];
    dst[gi] = v + OUT[gi];
  }
}

// =====================================================================
extern "C" void kernel_launch(void* const* d_in, const int* in_sizes, int n_in,
                              void* d_out, int out_size, void* d_ws, size_t ws_size,
                              hipStream_t stream) {
  const float* x        = (const float*)d_in[0];
  const float* br_g     = (const float*)d_in[1];
  const float* br_b     = (const float*)d_in[2];
  const float* lisa_w   = (const float*)d_in[3];
  const float* lisa_in  = (const float*)d_in[4];
  const float* lisa_ll  = (const float*)d_in[5];
  const float* lisa_lh  = (const float*)d_in[6];
  const float* mix_g    = (const float*)d_in[7];
  const float* mix_b    = (const float*)d_in[8];
  const float* convb_w  = (const float*)d_in[9];
  const float* convb_b  = (const float*)d_in[10];
  const float* gn_g     = (const float*)d_in[11];
  const float* gn_b     = (const float*)d_in[12];
  const float* convb_a  = (const float*)d_in[13];
  const float* q_w      = (const float*)d_in[14];
  const float* q_b      = (const float*)d_in[15];
  const float* q_a      = (const float*)d_in[16];
  const float* q_lng    = (const float*)d_in[17];
  const float* q_lnb    = (const float*)d_in[18];
  const float* k_w      = (const float*)d_in[19];
  const float* k_b      = (const float*)d_in[20];
  const float* k_a      = (const float*)d_in[21];
  const float* k_lng    = (const float*)d_in[22];
  const float* k_lnb    = (const float*)d_in[23];
  const float* v_w      = (const float*)d_in[24];
  const float* v_b      = (const float*)d_in[25];
  const float* v_a      = (const float*)d_in[26];
  const float* v_lng    = (const float*)d_in[27];
  const float* v_lnb    = (const float*)d_in[28];
  const float* proj_w   = (const float*)d_in[29];
  const float* proj_b   = (const float*)d_in[30];
  const float* proj_a   = (const float*)d_in[31];
  const float* proj_lng = (const float*)d_in[32];
  const float* proj_lnb = (const float*)d_in[33];

  float* ws    = (float*)d_ws;
  float* bsum  = ws + OFF_BSUM;
  float* tmp   = ws + OFF_TMP;   // b1, then y, then o_buf
  float* outp  = ws + OFF_OUT;
  h16*   Qh    = (h16*)(ws + OFF_Q);
  h16*   Kh    = (h16*)(ws + OFF_K);
  float* Vm    = ws + OFF_VM;
  h16*   Vt    = (h16*)(ws + OFF_VT);
  float* Pscr  = ws + OFF_PSC;
  h16*   Pp    = (h16*)(ws + OFF_PP);
  float* mu    = ws + OFF_MU;
  float* rstd  = ws + OFF_RSTD;
  float* gapr  = ws + OFF_GAPR;
  float* gapc  = ws + OFF_GAPC;
  float* Rbc   = ws + OFF_R;
  float* filt  = ws + OFF_FILT;
  float* pS    = ws + OFF_GS;
  float* pQ    = ws + OFF_GQ;
  float* gnst  = ws + OFF_GN;
  float* dst   = (float*)d_out;

  const int GB_NTF  = (NTF_ + 255) / 256;                 // 2016
  const int GB_NCTF = (int)((NCTF_ + 255) / 256);         // 129000
  const int GB_BCT  = (B_ * C_ * T_ + 255) / 256;         // 1000
  const int GB_BCF  = (B_ * C_ * F_ + 255) / 256;         // 129
  const int NT16    = (TF_ + 15) / 16;                    // 8063
  const int TT16    = (T_ + 15) / 16;                     // 63 valid t tiles
  const int dils[3] = {3, 5, 7};
  const float invscale = 1.0f / sqrtf((float)(E_ * F_));

  // ---------- stage 1: three LISA branches -> bsum ----------
  for (int i = 0; i < 3; ++i) {
    int d = dils[i];
    k_ln_stats<<<GB_NTF, 256, 0, stream>>>(x, mu, rstd);
    k_gapc<<<B_ * C_, 256, 0, stream>>>(x, mu, rstd, br_g + i * C_, br_b + i * C_, Rbc);
    k_filt<<<1, 64, 0, stream>>>(Rbc, lisa_w + (size_t)i * 12 * C_, filt);
    k_rowmean<<<GB_BCT, 256, 0, stream>>>(x, mu, rstd, br_g + i * C_, br_b + i * C_, gapr);
    k_lisa_h<<<GB_NCTF, 256, 0, stream>>>(x, mu, rstd, br_g + i * C_, br_b + i * C_,
                                          gapr, filt, lisa_in + i * C_, lisa_ll + i * C_,
                                          lisa_lh + i * C_, d, tmp);
    k_ln_stats<<<GB_NTF, 256, 0, stream>>>(tmp, mu, rstd);
    k_gapc<<<B_ * C_, 256, 0, stream>>>(tmp, mu, rstd, br_g + (i + 3) * C_,
                                        br_b + (i + 3) * C_, Rbc);
    k_filt<<<1, 64, 0, stream>>>(Rbc, lisa_w + (size_t)(i + 3) * 12 * C_, filt);
    k_colmean<<<GB_BCF, 256, 0, stream>>>(tmp, mu, rstd, br_g + (i + 3) * C_,
                                          br_b + (i + 3) * C_, gapc);
    k_lisa_v_acc<<<GB_NCTF, 256, 0, stream>>>(tmp, mu, rstd, br_g + (i + 3) * C_,
                                              br_b + (i + 3) * C_, gapc, filt,
                                              lisa_in + (i + 3) * C_, lisa_ll + (i + 3) * C_,
                                              lisa_lh + (i + 3) * C_, d, x,
                                              mix_g + i * C_, mix_b + i * C_, bsum,
                                              (i == 0) ? 1 : 0);
  }

  // ---------- stage 2: convb (WMMA) + per-batch norm + prelu -> output ----------
  k_gemm_cc<<<dim3(NT16, C_ / 16, B_), 32, 0, stream>>>(convb_w, bsum, convb_b, tmp,
                                                        0, nullptr);
  k_gn_part<<<B_ * C_, 256, 0, stream>>>(tmp, pS, pQ);
  k_gn_fin<<<B_, 64, 0, stream>>>(pS, pQ, gnst);
  k_gn_apply<<<GB_NCTF, 256, 0, stream>>>(tmp, gnst, gn_g, gn_b, convb_a, outp);

  // ---------- stage 3: head projections + row LN ----------
  {
    long long tq = (long long)H_ * B_ * TP_ * EFP_;
    int gq = (int)((tq + 255) / 256);
    k_headproj_h<<<gq, 256, 0, stream>>>(outp, q_w, q_b, q_a, Qh, E_, EFP_);
    k_headproj_h<<<gq, 256, 0, stream>>>(outp, k_w, k_b, k_a, Kh, E_, EFP_);
    long long tv = (long long)H_ * B_ * TP_ * DF_;
    int gv = (int)((tv + 255) / 256);
    k_headproj_f<<<gv, 256, 0, stream>>>(outp, v_w, v_b, v_a, Vm, D_, DF_);
    k_rownorm_h<<<H_ * B_ * T_, 256, 0, stream>>>(Qh, q_lng, q_lnb, E_, EFP_);
    k_rownorm_h<<<H_ * B_ * T_, 256, 0, stream>>>(Kh, k_lng, k_lnb, E_, EFP_);
    k_rownorm_f<<<H_ * B_ * T_, 256, 0, stream>>>(Vm, v_lng, v_lnb, D_, DF_);
    long long tt = (long long)H_ * B_ * DFP_ * TP_;
    int gt = (int)((tt + 255) / 256);
    k_v_transpose<<<gt, 256, 0, stream>>>(Vm, Vt);
  }

  // ---------- stage 4: attention (WMMA, f16 operands, branch-free loops) --------
  k_att_scores<<<dim3(TP_ / 32, TP_ / 16, H_ * B_), 32, 0, stream>>>(Qh, Kh, Pscr,
                                                                     invscale);
  k_softmax<<<H_ * B_ * TP_, 256, 0, stream>>>(Pscr, Pp);
  k_att_o<<<dim3((DF_ + 31) / 32, TT16, H_ * B_), 32, 0, stream>>>(Pp, Vt, tmp);

  // ---------- stage 5: proj (WMMA, fused prelu) + ln4dcf + residual ----------
  k_gemm_cc<<<dim3(NT16, C_ / 16, B_), 32, 0, stream>>>(proj_w, tmp, proj_b, bsum,
                                                        1, proj_a);
  k_lncf_final<<<B_ * T_, 256, 0, stream>>>(bsum, outp, proj_lng, proj_lnb, dst);
}
